// Mixture_Squashed_Gaussian_Actor_84439057039847
// MI455X (gfx1250) — matrix operand order repaired and checked by reading, not compiled, not run on hardware
//
#include <hip/hip_runtime.h>
#include <hip/hip_bf16.h>

// ---------------------------------------------------------------------------
// Mixture Squashed Gaussian Actor — fused 3-layer MLP + mixture head, gfx1250.
// GEMMs run on v_wmma_f32_16x16x32_f16 (f16 in, f32 accumulate).
// A-fragments are register-resident per stage; B streams from L2 in packed
// WMMA-fragment layout; two N-tiles in flight per loop for wmma density.
// ---------------------------------------------------------------------------

typedef _Float16 v16h __attribute__((ext_vector_type(16)));
typedef _Float16 v8h  __attribute__((ext_vector_type(8)));
typedef float    v8f  __attribute__((ext_vector_type(8)));

#define S_IN   376
#define H1     400
#define H2     300
#define NA     17
#define NM     5
#define OUTC   255   // 3*A*M

// padded tile geometry
#define K1T 12   // 384 = 12*32   (K of GEMM1, pads 376)
#define N1T 25   // 400 = 25*16
#define K2T 13   // 416 = 13*32   (pads 400)
#define N2T 19   // 304 = 19*16   (pads 300)
#define K3T 10   // 320 = 10*32   (pads 300)
#define N3T 16   // 256 = 16*16   (pads 255)

// packed weight sizes (halves)
#define W1P_CNT (K1T * N1T * 512)   // 153600
#define W2P_CNT (K2T * N2T * 512)   // 126464
#define W3P_CNT (K3T * N3T * 512)   //  81920

// LDS strides (elements), padded to dodge bank conflicts, 16B-aligned rows
#define SA_STRIDE  392   // state tile, f16  (128*392*2 = 100352 B)
#define H1_STRIDE  424   // h1 tile,   f16  (128*424*2 = 108544 B)
#define H2_STRIDE  328   // h2 tile,   f16  (128*328*2 =  83968 B)
#define OUT_STRIDE 257   // out tile,  f32  (128*257*4 = 131584 B)

#define REGION_B_OFF 100352
#define SMEM_BYTES   (100352 + 131584)   // 231936 <= 320KB/WGP

#define TM 128          // rows per block
#define EPIS_MIN 4.53999297624848e-05f
#define EPIS_MAX 7.38905609893065f
#define LOG2PI   1.83787706640934548f

// ---------------------------------------------------------------------------
// Weight packer: f32 row-major [Ksrc x Nsrc]  ->  f16 WMMA B-fragment stream.
// B 32x16 fragment (ISA 7.12.2): lane l holds column n=l&15, halves i=0..15
// hold K = kt*32 + (i&7) + 16*(i>=8) + 8*(l>=16).  Packed flat so a lane's 16 halves are
// contiguous:  dst[((kt*NT+nt)*32 + lane)*16 + i].
// ---------------------------------------------------------------------------
__global__ void pack_w_kernel(const float* __restrict__ W, int Ksrc, int Nsrc,
                              int NT, _Float16* __restrict__ dst, int count) {
    int tid = blockIdx.x * 256 + threadIdx.x;
    if (tid >= count) return;
    int i    = tid & 15;
    int lane = (tid >> 4) & 31;
    int pair = tid >> 9;
    int nt   = pair % NT;
    int kt   = pair / NT;
    int n = nt * 16 + (lane & 15);
    int k = kt * 32 + (i & 7) + ((i & 8) ? 16 : 0) + ((lane & 16) ? 8 : 0);
    float v = (k < Ksrc && n < Nsrc) ? W[(size_t)k * Nsrc + n] : 0.0f;
    dst[tid] = (_Float16)v;
}

// A-fragment (ISA 7.12.2, 16-bit A 16x32): lane l holds row m=l&15.
// For l<16 : halves 0-7 = K kbase+0..7,  halves 8-15 = K kbase+16..23
// For l>=16: halves 0-7 = K kbase+8..15, halves 8-15 = K kbase+24..31
// rowPtr must already include   row*stride + kbase + (l>=16 ? 8 : 0).
static __device__ __forceinline__ v16h load_a_frag(const _Float16* rowPtr) {
    v8h lo = *(const v8h*)(rowPtr);
    v8h hi = *(const v8h*)(rowPtr + 16);
    return __builtin_shufflevector(lo, hi, 0, 1, 2, 3, 4, 5, 6, 7,
                                           8, 9, 10, 11, 12, 13, 14, 15);
}

// ---------------------------------------------------------------------------
// One GEMM stage: A-tile (16 rows x KT*32) register-resident, B streamed from
// packed global weights, 2 N-tiles in flight. Epi(nt, acc) handles bias +
// activation + LDS store.
// ---------------------------------------------------------------------------
template <int KT, int NT, class Epi>
static __device__ __forceinline__ void gemm_stage(const _Float16* aRow,
                                                  const _Float16* __restrict__ Wp,
                                                  int lane, Epi epi) {
    v16h aF[KT];
#pragma unroll
    for (int kt = 0; kt < KT; ++kt)
        aF[kt] = load_a_frag(aRow + kt * 32);

    int nt = 0;
    for (; nt + 1 < NT; nt += 2) {
        // prefetch the first line of the nt+2 column block (L2-resident hint)
        int ntp = (nt + 2 < NT) ? nt + 2 : nt;
        __builtin_prefetch(Wp + (((size_t)ntp) * 32 + lane) * 16, 0, 1);
        v8f acc0 = {}, acc1 = {};
#pragma unroll
        for (int kt = 0; kt < KT; ++kt) {
            const _Float16* bp0 = Wp + (((size_t)kt * NT + nt) * 32 + lane) * 16;
            v16h b0 = *(const v16h*)bp0;
            v16h b1 = *(const v16h*)(bp0 + 512);   // nt+1: +32 lanes * 16 halves
            acc0 = __builtin_amdgcn_wmma_f32_16x16x32_f16(
                false, aF[kt], false, b0, (short)0, acc0, false, false);
            acc1 = __builtin_amdgcn_wmma_f32_16x16x32_f16(
                false, aF[kt], false, b1, (short)0, acc1, false, false);
        }
        epi(nt, acc0);
        epi(nt + 1, acc1);
    }
    if (nt < NT) {   // odd NT tail
        v8f acc = {};
#pragma unroll
        for (int kt = 0; kt < KT; ++kt) {
            v16h b = *(const v16h*)(Wp + (((size_t)kt * NT + nt) * 32 + lane) * 16);
            acc = __builtin_amdgcn_wmma_f32_16x16x32_f16(
                false, aF[kt], false, b, (short)0, acc, false, false);
        }
        epi(nt, acc);
    }
}

// ---------------------------------------------------------------------------
// Fused actor kernel. 256 threads = 8 wave32; block owns 128 rows.
// Wave w owns rows [w*16, w*16+16): all A-fragments are wave-private, so only
// region-reuse barriers are needed between stages.
// ---------------------------------------------------------------------------
__launch_bounds__(256)
__global__ void actor_fused_kernel(const float* __restrict__ state,
                                   const float* __restrict__ eps,
                                   const float* __restrict__ b1,
                                   const float* __restrict__ b2,
                                   const float* __restrict__ b3,
                                   const _Float16* __restrict__ W1p,
                                   const _Float16* __restrict__ W2p,
                                   const _Float16* __restrict__ W3p,
                                   float* __restrict__ out, int B) {
    extern __shared__ char smem[];
    _Float16* sA   = (_Float16*)smem;                    // region A: state tile
    _Float16* sH2  = (_Float16*)smem;                    // region A: h2 (reuse)
    _Float16* sH1  = (_Float16*)(smem + REGION_B_OFF);   // region B: h1
    float*    sOut = (float*)   (smem + REGION_B_OFF);   // region B: out (reuse)

    const int tid    = threadIdx.x;
    const int lane   = tid & 31;
    const int w      = tid >> 5;
    const int hiHalf = (lane >= 16) ? 1 : 0;
    const int col16  = lane & 15;
    const int rBase  = blockIdx.x * TM;
    const int rowA   = w * 16 + col16;            // A-fragment source row (local)
    const int rlBase = w * 16 + hiHalf * 8;       // C/D fragment row base (local)

    // ---- stage 0: state (f32 global) -> f16 LDS tile, K zero-padded --------
    for (int idx = tid; idx < TM * SA_STRIDE; idx += 256) {
        int r = idx / SA_STRIDE, k = idx - r * SA_STRIDE;
        float v = (k < S_IN) ? state[(size_t)(rBase + r) * S_IN + k] : 0.0f;
        sA[idx] = (_Float16)v;
    }
    // zero h1 pad columns (K-pad for GEMM2): cols 400..423
    for (int idx = tid; idx < TM * 24; idx += 256) {
        int r = idx / 24, c = H1 + (idx - r * 24);
        sH1[r * H1_STRIDE + c] = (_Float16)0.0f;
    }
    __syncthreads();

    // ---- GEMM1: h1 = relu(state @ W1 + b1) ---------------------------------
    gemm_stage<K1T, N1T>(
        sA + rowA * SA_STRIDE + hiHalf * 8, W1p, lane,
        [&](int nt, v8f acc) {
            int col = nt * 16 + col16;
            float bias = b1[col];
#pragma unroll
            for (int j = 0; j < 8; ++j) {
                float v = acc[j] + bias;
                sH1[(rlBase + j) * H1_STRIDE + col] = (_Float16)(v > 0.0f ? v : 0.0f);
            }
        });
    __syncthreads();   // all reads of sA done -> region A reusable as h2

    // zero h2 pad columns (K-pad for GEMM3): cols 304..327
    for (int idx = tid; idx < TM * 24; idx += 256) {
        int r = idx / 24, c = 304 + (idx - r * 24);
        sH2[r * H2_STRIDE + c] = (_Float16)0.0f;
    }

    // ---- GEMM2: h2 = relu(h1 @ W2 + b2) ------------------------------------
    gemm_stage<K2T, N2T>(
        sH1 + rowA * H1_STRIDE + hiHalf * 8, W2p, lane,
        [&](int nt, v8f acc) {
            int col = nt * 16 + col16;
            float bias = (col < H2) ? b2[col] : 0.0f;
#pragma unroll
            for (int j = 0; j < 8; ++j) {
                float v = acc[j] + bias;
                sH2[(rlBase + j) * H2_STRIDE + col] = (_Float16)(v > 0.0f ? v : 0.0f);
            }
        });
    __syncthreads();   // all reads of sH1 done -> region B reusable as sOut

    // ---- GEMM3: raw = h2 @ W3 + b3 (f32 kept in LDS, tanh in epilogue) -----
    gemm_stage<K3T, N3T>(
        sH2 + rowA * H2_STRIDE + hiHalf * 8, W3p, lane,
        [&](int nt, v8f acc) {
            int col = nt * 16 + col16;
            float bias = (col < OUTC) ? b3[col] : 0.0f;
#pragma unroll
            for (int j = 0; j < 8; ++j)
                sOut[(rlBase + j) * OUT_STRIDE + col] = acc[j] + bias;
        });
    __syncthreads();

    // ---- mixture head + rsample + tanh squash + log-probs (f32) ------------
    if (tid < TM) {
        const float* o = sOut + tid * OUT_STRIDE;
        const int row = rBase + tid;
        float lp_t_acc = 0.0f, lp_e_acc = 0.0f, corr = 0.0f;
        for (int a = 0; a < NA; ++a) {
            float mw[NM], mm[NM], sd[NM], wexp[NM];
            float mx = -3.4e38f;
            for (int m = 0; m < NM; ++m) {
                mw[m] = tanhf(o[a * NM + m]);
                mm[m] = tanhf(o[NA * NM + a * NM + m]);
                float lms = tanhf(o[2 * NA * NM + a * NM + m]);
                lms = fminf(fmaxf(lms, -10.0f), 2.0f);
                sd[m] = expf(lms);
                mx = fmaxf(mx, mw[m]);
            }
            float wsum = 0.0f;
            for (int m = 0; m < NM; ++m) { wexp[m] = expf(mw[m] - mx); wsum += wexp[m]; }
            float inv = 1.0f / wsum;
            float mean = 0.0f, alea = 0.0f;
            for (int m = 0; m < NM; ++m) {
                float wgt = wexp[m] * inv;
                mean += wgt * mm[m];
                alea += wgt * sd[m];
            }
            float epis = 0.0f;
            for (int m = 0; m < NM; ++m) {
                float d = mm[m] - mean;
                epis += wexp[m] * inv * d * d;
            }
            epis = fminf(fmaxf(epis, EPIS_MIN), EPIS_MAX);
            float total  = alea + epis;
            float e      = eps[(size_t)row * NA + a];
            float sample = mean + total * e;        // (sample-mean)/total == e
            float t      = tanhf(sample);
            lp_t_acc += e * e + 2.0f * logf(total);
            float ze = (total * e) / epis;
            lp_e_acc += ze * ze + 2.0f * logf(epis);
            corr += logf(1.0f - t * t + 1e-6f);
            out[(size_t)row * NA + a] = t;
        }
        const float C = 0.5f * (float)NA * LOG2PI;
        out[(size_t)B * NA + row]       = (-0.5f * lp_t_acc - C) - corr;
        out[(size_t)B * (NA + 1) + row] = (-0.5f * lp_e_acc - C) - corr;
    }
}

// ---------------------------------------------------------------------------
extern "C" void kernel_launch(void* const* d_in, const int* in_sizes, int n_in,
                              void* d_out, int out_size, void* d_ws, size_t ws_size,
                              hipStream_t stream) {
    const float* state = (const float*)d_in[0];
    const float* eps   = (const float*)d_in[1];
    const float* W1    = (const float*)d_in[2];
    const float* b1    = (const float*)d_in[3];
    const float* W2    = (const float*)d_in[4];
    const float* b2    = (const float*)d_in[5];
    const float* W3    = (const float*)d_in[6];
    const float* b3    = (const float*)d_in[7];
    float* out = (float*)d_out;

    const int B = in_sizes[0] / S_IN;   // 131072

    char* ws = (char*)d_ws;
    _Float16* W1p = (_Float16*)ws;                                     // 307200 B
    _Float16* W2p = (_Float16*)(ws + (size_t)W1P_CNT * 2);             // 252928 B
    _Float16* W3p = (_Float16*)(ws + (size_t)(W1P_CNT + W2P_CNT) * 2); // 163840 B

    pack_w_kernel<<<(W1P_CNT + 255) / 256, 256, 0, stream>>>(W1, S_IN, H1, N1T, W1p, W1P_CNT);
    pack_w_kernel<<<(W2P_CNT + 255) / 256, 256, 0, stream>>>(W2, H1, H2, N2T, W2p, W2P_CNT);
    pack_w_kernel<<<(W3P_CNT + 255) / 256, 256, 0, stream>>>(W3, H2, OUTC, N3T, W3p, W3P_CNT);

    actor_fused_kernel<<<B / TM, 256, SMEM_BYTES, stream>>>(
        state, eps, b1, b2, b3, W1p, W2p, W3p, out, B);
}
